// KernelSetConv_80668075753605
// MI455X (gfx1250) — matrix operand order repaired and checked by reading, not compiled, not run on hardware
//
#include <hip/hip_runtime.h>
#include <math.h>

// ---------------- CDNA5 WMMA types ----------------
typedef _Float16 f16;
typedef __attribute__((ext_vector_type(16))) _Float16 v16h;
typedef __attribute__((ext_vector_type(8)))  _Float16 v8h;   // 16B packed half loads
typedef __attribute__((ext_vector_type(8)))  float    v8f;
typedef __attribute__((ext_vector_type(4)))  float    f32x4; // 16B packed float loads

#define LDIM   64
#define MDIM   3
#define FDIM   32
#define EDIM   8
#define NPERM  6
#define NROWS  (NPERM * LDIM)      // 384 support rows (p-major: row = p*64 + l)
#define HALF_PI 1.57079632679489661923f
#define EPSV    1e-8f

// itertools.permutations(range(3)) order
__device__ const int PERM3[NPERM][MDIM] = {
    {0,1,2},{0,2,1},{1,0,2},{1,2,0},{2,0,1},{2,1,0}};

// ---------------- workspace layout (bytes, 256-aligned blocks) ----------------
static constexpr size_t OFF_XSUP = 0;        // (384,96) f16  = 73728
static constexpr size_t OFF_ESUP = 73728;    // (384,32) f16  = 24576
static constexpr size_t OFF_XCH  = 98304;    // ( 64,32) f16  =  4096
static constexpr size_t OFF_SSQ  = 102400;   // (384) f32
static constexpr size_t OFF_ESQ  = 103936;
static constexpr size_t OFF_XCSQ = 105472;   // (64) f32
static constexpr size_t OFF_ASQ  = 105728;
static constexpr size_t OFF_LSQ  = 107264;
static constexpr size_t OFF_ANGS = 108800;   // (384,3) f32
static constexpr size_t OFF_LENS = 113408;   // end = 118016 bytes

// ---------------- fast math helpers ----------------
// Odd polynomial for atan on [-1,1] (Hastings, deg 13, ~1.3e-2 abs err at |t|=1).
__device__ __forceinline__ float atan_poly(float t) {
    float s = t * t;
    float p =            0.0429096138f;
    p = fmaf(p, s, -0.0752896400f);
    p = fmaf(p, s,  0.1065626393f);
    p = fmaf(p, s, -0.1420889944f);
    p = fmaf(p, s,  0.1999355085f);
    p = fmaf(p, s, -0.3333314528f);
    return fmaf(p * s, t, t);
}

// atan(1/d) via fast v_rcp_f32, no IEEE division chains.
// |d|>=1 : atan(rcp(d)) directly (|arg|<=1)
// |d|< 1 : sign(d)*pi/2 - atan(d)      (d==+0 -> +pi/2, matching arctan(inf))
__device__ __forceinline__ float atan_inv(float d) {
    float ad  = fabsf(d);
    float t   = __builtin_amdgcn_rcpf(d);      // fast hw reciprocal
    bool  big = ad >= 1.0f;
    float x   = big ? t : d;
    float pp  = atan_poly(x);
    float alt = copysignf(HALF_PI, d) - pp;
    return big ? pp : alt;
}

// ---------------- WMMA operand loaders ----------------
// A-matrix 16x32 f16 layout (ISA 7.12.2): lane m = lane&15, hi = lane>>4.
// halves i : k = 16*hi + i (contiguous, matching B).
__device__ __forceinline__ v16h load_a_chunk(const f16* rowk, int hi) {
    v8h lo = *(const v8h*)(rowk + hi * 16);
    v8h hp = *(const v8h*)(rowk + hi * 16 + 8);
    v16h a;
#pragma unroll
    for (int i = 0; i < 8; ++i) { a[i] = lo[i]; a[i + 8] = hp[i]; }
    return a;
}

// B-matrix 32x16 f16: lane n = lane&15, lanes 0-15 hold K=0..15,
// lanes 16-31 hold K=16..31, contiguous. Also accumulates sum of squares.
__device__ __forceinline__ v16h load_b_f32(const float* base, float& sq) {
    f32x4 q0 = ((const f32x4*)base)[0];
    f32x4 q1 = ((const f32x4*)base)[1];
    f32x4 q2 = ((const f32x4*)base)[2];
    f32x4 q3 = ((const f32x4*)base)[3];
    v16h b;
#pragma unroll
    for (int i = 0; i < 4; ++i) {
        b[i]      = (f16)q0[i]; sq += q0[i] * q0[i];
        b[i + 4]  = (f16)q1[i]; sq += q1[i] * q1[i];
        b[i + 8]  = (f16)q2[i]; sq += q2[i] * q2[i];
        b[i + 12] = (f16)q3[i]; sq += q3[i] * q3[i];
    }
    return b;
}

__device__ __forceinline__ v8f wmma_f16(v16h a, v16h b, v8f c) {
    return __builtin_amdgcn_wmma_f32_16x16x32_f16(
        /*neg_a=*/false, a, /*neg_b=*/false, b,
        /*c_mod=*/(short)0, c, /*reuse_a=*/false, /*reuse_b=*/false);
}

// ---------------- kernel 1: permute + f16-cast support side into ws ----------------
__global__ __launch_bounds__(128) void ksc_precompute(
    const float* __restrict__ xc_in, const float* __restrict__ xs_in,
    const float* __restrict__ es_in, const float* __restrict__ ps_in,
    unsigned char* __restrict__ ws)
{
    int t = blockIdx.x * blockDim.x + threadIdx.x;
    f16*   xsup   = (f16*)(ws + OFF_XSUP);
    f16*   esup   = (f16*)(ws + OFF_ESUP);
    f16*   xch    = (f16*)(ws + OFF_XCH);
    float* ssqA   = (float*)(ws + OFF_SSQ);
    float* esqA   = (float*)(ws + OFF_ESQ);
    float* xcsqA  = (float*)(ws + OFF_XCSQ);
    float* asqA   = (float*)(ws + OFF_ASQ);
    float* lsqA   = (float*)(ws + OFF_LSQ);
    float* angS   = (float*)(ws + OFF_ANGS);
    float* lenS   = (float*)(ws + OFF_LENS);

    if (t < NROWS) {
        int p = t >> 6, l = t & 63;
        float ssq = 0.f, esq = 0.f;
        for (int m = 0; m < MDIM; ++m) {
            int src = l * MDIM + PERM3[p][m];
            for (int f = 0; f < FDIM; ++f) {
                float v = xs_in[src * FDIM + f];
                xsup[(size_t)t * 96 + m * FDIM + f] = (f16)v;
                ssq += v * v;
            }
            for (int e = 0; e < EDIM; ++e) {
                float v = es_in[src * EDIM + e];
                esup[(size_t)t * 32 + m * EDIM + e] = (f16)v;
                esq += v * v;
            }
        }
        for (int k = 24; k < 32; ++k) esup[(size_t)t * 32 + k] = (f16)0.f;
        ssqA[t] = ssq; esqA[t] = esq;

        float ps[MDIM][3], len[MDIM], ang[MDIM];
        for (int m = 0; m < MDIM; ++m)
            for (int d = 0; d < 3; ++d)
                ps[m][d] = ps_in[(size_t)(l * MDIM + PERM3[p][m]) * 3 + d];
        for (int m = 0; m < MDIM; ++m)
            len[m] = sqrtf(ps[m][0]*ps[m][0] + ps[m][1]*ps[m][1] + ps[m][2]*ps[m][2]);
        for (int m = 0; m < MDIM; ++m) {
            int q = (m + MDIM - 1) % MDIM;   // roll by +1 along m
            float num = ps[q][0]*ps[m][0] + ps[q][1]*ps[m][1] + ps[q][2]*ps[m][2];
            ang[m] = num * __builtin_amdgcn_rcpf(fmaxf(len[q] * len[m], EPSV));
        }
        float asq = 0.f, lsq = 0.f;
        for (int m = 0; m < MDIM; ++m) {
            angS[t * 3 + m] = ang[m]; lenS[t * 3 + m] = len[m];
            asq += ang[m] * ang[m];   lsq += len[m] * len[m];
        }
        asqA[t] = asq; lsqA[t] = lsq;
    }
    if (t < LDIM) {   // x_center is (L,1,F)
        float s = 0.f;
        for (int f = 0; f < FDIM; ++f) {
            float v = xc_in[(size_t)t * FDIM + f];
            xch[(size_t)t * 32 + f] = (f16)v;
            s += v * v;
        }
        xcsqA[t] = s;
    }
}

// ---------------- kernel 2: fused WMMA scoring ----------------
// One wave (32 lanes) owns 16 output columns (n). 8 waves / block.
// Requires N % 16 == 0 (true: 100000).
__global__ __launch_bounds__(256) void ksc_main(
    const float* __restrict__ xf, const float* __restrict__ pf,
    const float* __restrict__ xn, const float* __restrict__ pn,
    const float* __restrict__ en, const unsigned char* __restrict__ ws,
    float* __restrict__ out, int N)
{
    const int lane = threadIdx.x & 31;
    const int wave = threadIdx.x >> 5;
    const int n0   = (blockIdx.x * 8 + wave) * 16;
    if (n0 >= N) return;                     // wave-uniform exit (EXEC stays full)
    const int hi  = lane >> 4;
    const int col = n0 + (lane & 15);

    const f16*   xsup  = (const f16*)(ws + OFF_XSUP);
    const f16*   esup  = (const f16*)(ws + OFF_ESUP);
    const f16*   xch   = (const f16*)(ws + OFF_XCH);
    const float* ssqA  = (const float*)(ws + OFF_SSQ);
    const float* esqA  = (const float*)(ws + OFF_ESQ);
    const float* xcsqA = (const float*)(ws + OFF_XCSQ);
    const float* asqA  = (const float*)(ws + OFF_ASQ);
    const float* lsqA  = (const float*)(ws + OFF_LSQ);
    const float* angS  = (const float*)(ws + OFF_ANGS);
    const float* lenS  = (const float*)(ws + OFF_LENS);

    // ---- per-lane B operands (neighbor side), kept in VGPRs all kernel ----
    v16h bX[3]; float nsq = 0.f;
    {
        const float* xcol = xn + (size_t)col * 96;
#pragma unroll
        for (int s = 0; s < 3; ++s)
            bX[s] = load_b_f32(xcol + s * 32 + hi * 16, nsq);
        nsq += __shfl_xor(nsq, 16, 32);      // combine K halves
    }
    v16h bE; float esq = 0.f;
    {
        const float* ecol = en + (size_t)col * 24 + hi * 16;
        f32x4 q0 = ((const f32x4*)ecol)[0];
        f32x4 q1 = ((const f32x4*)ecol)[1];
        f32x4 q2 = {0.f,0.f,0.f,0.f}, q3 = {0.f,0.f,0.f,0.f};
        if (hi == 0) { q2 = ((const f32x4*)ecol)[2]; q3 = ((const f32x4*)ecol)[3]; }
#pragma unroll
        for (int i = 0; i < 4; ++i) {
            bE[i]      = (f16)q0[i]; esq += q0[i]*q0[i];
            bE[i + 4]  = (f16)q1[i]; esq += q1[i]*q1[i];
            bE[i + 8]  = (f16)q2[i]; esq += q2[i]*q2[i];
            bE[i + 12] = (f16)q3[i]; esq += q3[i]*q3[i];
        }
        esq += __shfl_xor(esq, 16, 32);
    }
    v16h bC; float xfsq = 0.f;
    {
        bC = load_b_f32(xf + (size_t)col * 32 + hi * 16, xfsq);
        xfsq += __shfl_xor(xfsq, 16, 32);
    }
    // ---- K=3 neighbor angle/length vectors (VALU) ----
    float prel[MDIM][3], lenn[MDIM], angn[MDIM];
#pragma unroll
    for (int m = 0; m < MDIM; ++m)
#pragma unroll
        for (int d = 0; d < 3; ++d)
            prel[m][d] = pn[(size_t)col * 9 + m * 3 + d] - pf[(size_t)col * 3 + d];
#pragma unroll
    for (int m = 0; m < MDIM; ++m)
        lenn[m] = sqrtf(prel[m][0]*prel[m][0] + prel[m][1]*prel[m][1] + prel[m][2]*prel[m][2]);
#pragma unroll
    for (int m = 0; m < MDIM; ++m) {
        int q = (m + MDIM - 1) % MDIM;
        float num = prel[q][0]*prel[m][0] + prel[q][1]*prel[m][1] + prel[q][2]*prel[m][2];
        angn[m] = num * __builtin_amdgcn_rcpf(fmaxf(lenn[q] * lenn[m], EPSV));
    }
    float asq_n = angn[0]*angn[0] + angn[1]*angn[1] + angn[2]*angn[2];
    float lsq_n = lenn[0]*lenn[0] + lenn[1]*lenn[1] + lenn[2]*lenn[2];

    // ---- 4 l-tiles of 16 rows; inner loop over 6 permutations ----
    for (int lt = 0; lt < 4; ++lt) {
        // center GEMM (rows lt*16..lt*16+15, K=32)
        v8f cacc = {};
        {
            v16h a = load_a_chunk(xch + (size_t)(lt * 16 + (lane & 15)) * 32, hi);
            cacc = wmma_f16(a, bC, cacc);
        }
        // running argmin over support distance; angle/length deferred to epilogue
        float bkey[8], bde[8];
        int   brr[8];
#pragma unroll
        for (int r = 0; r < 8; ++r) { bkey[r] = 3.0e38f; bde[r] = 0.f; brr[r] = 0; }

        for (int p = 0; p < NPERM; ++p) {
            const int rowbase = p * 64 + lt * 16;
            const f16* arow = xsup + (size_t)(rowbase + (lane & 15)) * 96;
            v8f acc = {};
#pragma unroll
            for (int s = 0; s < 3; ++s)
                acc = wmma_f16(load_a_chunk(arow + s * 32, hi), bX[s], acc);
            v8f eacc = {};
            eacc = wmma_f16(load_a_chunk(esup + (size_t)(rowbase + (lane & 15)) * 32, hi),
                            bE, eacc);
#pragma unroll
            for (int r = 0; r < 8; ++r) {
                const int rr = rowbase + r + 8 * hi;         // C/D row: M = r + 8*hi
                float ds = fmaf(-2.f, acc[r],  nsq + ssqA[rr]);
                float de = fmaf(-2.f, eacc[r], esq + esqA[rr]);
                bool u = ds < bkey[r];                       // strict <: first max wins
                bkey[r] = u ? ds : bkey[r];
                bde[r]  = u ? de : bde[r];
                brr[r]  = u ? rr : brr[r];
            }
        }
#pragma unroll
        for (int r = 0; r < 8; ++r) {
            const int row = lt * 16 + r + 8 * hi;
            const int rb  = brr[r];                          // winning support row
            // angle/length quadratics only for the winner (tables are L0-hot)
            float da = asq_n + asqA[rb]
                     - 2.f * (angn[0]*angS[rb*3] + angn[1]*angS[rb*3+1] + angn[2]*angS[rb*3+2]);
            float dl = lsq_n + lsqA[rb]
                     - 2.f * (lenn[0]*lenS[rb*3] + lenn[1]*lenS[rb*3+1] + lenn[2]*lenS[rb*3+2]);
            float dc = xfsq + xcsqA[row] - 2.f * cacc[r];
            float a0 = atan_inv(dl)      - HALF_PI;   // length score
            float a1 = atan_inv(da)      - HALF_PI;   // angle score
            float a2 = atan_inv(bkey[r]) - HALF_PI;   // support score
            float a3 = atan_inv(dc)      - HALF_PI;   // center score
            float a4 = atan_inv(bde[r])  - HALF_PI;   // edge score
            float h  = a0*a0 + a1*a1 + a2*a2 + a3*a3 + a4*a4;
            out[(size_t)row * N + col] = atan_inv(h);
        }
    }
}

// ---------------- host launcher ----------------
extern "C" void kernel_launch(void* const* d_in, const int* in_sizes, int n_in,
                              void* d_out, int out_size, void* d_ws, size_t ws_size,
                              hipStream_t stream) {
    const float* x_focal    = (const float*)d_in[0];
    const float* p_focal    = (const float*)d_in[1];
    const float* x_neighbor = (const float*)d_in[2];
    const float* p_neighbor = (const float*)d_in[3];
    const float* edge_nei   = (const float*)d_in[4];
    const float* x_center   = (const float*)d_in[5];
    const float* x_support  = (const float*)d_in[6];
    const float* edge_sup   = (const float*)d_in[7];
    const float* p_support  = (const float*)d_in[8];
    const int N = in_sizes[0] / FDIM;        // x_focal is (N, 32)

    unsigned char* ws = (unsigned char*)d_ws;
    ksc_precompute<<<3, 128, 0, stream>>>(x_center, x_support, edge_sup, p_support, ws);

    const int ngroups = (N + 15) / 16;       // 16 columns per wave (N % 16 == 0)
    const int nblocks = (ngroups + 7) / 8;   // 8 waves per 256-thread block
    ksc_main<<<nblocks, 256, 0, stream>>>(x_focal, p_focal, x_neighbor, p_neighbor,
                                          edge_nei, ws, (float*)d_out, N);
}
